// MaskedAutoregressiveFlow_1941325218523
// MI455X (gfx1250) — matrix operand order repaired and checked by reading, not compiled, not run on hardware
//
#include <hip/hip_runtime.h>
#include <hip/hip_bf16.h>

// CDNA5 / gfx1250 WMMA types (probe-verified builtin signature)
typedef __attribute__((ext_vector_type(16))) _Float16 v16h;
typedef __attribute__((ext_vector_type(8)))  float    v8f;

#define MAF_CLAMP 10.0f
#define LOG2E 1.4426950408889634f

__device__ __forceinline__ float fast_tanh(float x) {
    // tanh(x) = 1 - 2/(exp(2x)+1), via v_exp_f32 + v_rcp_f32 (trans ops)
    float e = __builtin_amdgcn_exp2f(x * (2.0f * LOG2E));
    return 1.0f - 2.0f * __builtin_amdgcn_rcpf(e + 1.0f);
}

__device__ __forceinline__ float scrub_nonfinite(float v) {
    // reference: where(isnan|isinf, 0, v)
    return (v == v && fabsf(v) <= 3.0e38f) ? v : 0.0f;
}

// One workgroup = 16 batch rows, 8 wave32.
// Wave w owns hidden units [w*128, w*128+128) as 8 WMMA N-tiles of 16.
__global__ __launch_bounds__(256, 1)
void maf_sample_kernel(const float* __restrict__ z,
                       const float* __restrict__ W1,
                       const float* __restrict__ b1,
                       const float* __restrict__ W2,
                       const float* __restrict__ b2,
                       float* __restrict__ out,
                       int B)
{
    __shared__ _Float16 xh[16][64];      // x tile, f16 (WMMA A operand source)
    __shared__ float    xf[16][64];      // x tile, f32 (exact output)
    __shared__ float    zf[16][64];      // z tile
    __shared__ float    red[2][16][8];   // per-wave reduced partials: [mu/alpha][row][wave]
    __shared__ float    ldacc[16];       // log_det accumulators

    const int tid   = threadIdx.x;
    const int lane  = tid & 31;
    const int wave  = tid >> 5;
    const int hi    = lane >> 4;     // lane group (0: lanes 0-15, 1: lanes 16-31)
    const int lsub  = lane & 15;
    const int b0    = blockIdx.x * 16;
    const int nbase = wave * 128;

    // ---- init tile state ----
    for (int idx = tid; idx < 16 * 64; idx += 256) {
        ((_Float16*)xh)[idx] = (_Float16)0.0f;
        ((float*)xf)[idx]    = 0.0f;
        ((float*)zf)[idx]    = z[(size_t)b0 * 64 + idx];
    }
    if (tid < 16) ldacc[tid] = 0.0f;

    // ---- one-time: masked W1^T fragments -> f16, pinned in VGPRs ----
    // B-operand (K x N = 64 x 16 per tile, two K=32 steps). B k order used here (contiguous):
    // lanes 0-15 hold K = k0+0..15, lanes 16-31 hold K = k0+16..31, N = lane&15.
    v16h  bf[8][2];
    float b1v[8];
    int   hv[8], m0v[8];
    #pragma unroll
    for (int t = 0; t < 8; ++t) {
        const int n  = nbase + t * 16 + lsub;   // hidden unit index
        const int m0 = n % 63;                  // mask degree: W1m[n][d] = W1[n][d]*(d<=m0)
        hv[t]  = n;
        m0v[t] = m0;
        b1v[t] = b1[n];
        #pragma unroll
        for (int s = 0; s < 2; ++s) {
            const int kb = s * 32 + (hi << 4);
            v16h b;
            #pragma unroll
            for (int j = 0; j < 16; ++j) {
                const int d = kb + j;
                const float w = (d <= m0) ? W1[(size_t)n * 64 + d] : 0.0f;
                b[j] = (_Float16)w;
            }
            bf[t][s] = b;
        }
    }
    __syncthreads();

    // ---- 64 sequential autoregressive steps ----
    for (int i = 0; i < 64; ++i) {
        // A-operand fragments from xh. A 16x32 k order used here (contiguous):
        // lanes 0-15: K = 0..15; lanes 16-31: K = 16..31 (matches B fragment order).
        v16h a0, a1;
        {
            const int m    = lsub;
            const int koff = hi << 4;  // 0 or 16
            #pragma unroll
            for (int j = 0; j < 8; ++j) {
                a0[j]     = xh[m][koff + j];
                a0[8 + j] = xh[m][koff + 8 + j];
                a1[j]     = xh[m][32 + koff + j];
                a1[8 + j] = xh[m][32 + koff + 8 + j];
            }
        }

        if (lane == 0 && i < 63) {  // keep next step's W2 rows warm near the WGP
            __builtin_prefetch(&W2[(size_t)(i + 1) * 1024 + nbase], 0, 3);
            __builtin_prefetch(&W2[(size_t)(64 + i + 1) * 1024 + nbase], 0, 3);
        }

        const int thr_mu = i >> 1;
        const int thr_al = 32 + (i >> 1);
        const float* __restrict__ w2mu_row = W2 + (size_t)i * 1024;
        const float* __restrict__ w2al_row = W2 + (size_t)(64 + i) * 1024;

        float pm[8], pa[8];
        #pragma unroll
        for (int e = 0; e < 8; ++e) { pm[e] = 0.0f; pa[e] = 0.0f; }

        #pragma unroll
        for (int t = 0; t < 8; ++t) {
            // S(16x16) = Xtile(16x64) * W1m^T tile(64x16), f32 accumulate on matrix cores
            v8f c = {};
            c = __builtin_amdgcn_wmma_f32_16x16x32_f16(false, a0, false, bf[t][0],
                                                       (short)0, c, false, false);
            c = __builtin_amdgcn_wmma_f32_16x16x32_f16(false, a1, false, bf[t][1],
                                                       (short)0, c, false, false);

            // masked W2 columns: UNCONDITIONAL load (always in-bounds) + v_cndmask,
            // avoids exec-mask branches around every load.
            const float w2m = w2mu_row[hv[t]];
            const float w2a = w2al_row[hv[t]];
            const float wmu = (m0v[t] < thr_mu) ? w2m : 0.0f;
            const float wal = (m0v[t] < thr_al) ? w2a : 0.0f;

            // fused tanh + two masked dot-product partials (VALU, co-executes with WMMA)
            #pragma unroll
            for (int e = 0; e < 8; ++e) {
                const float tv = fast_tanh(c[e] + b1v[t]);
                pm[e] = fmaf(tv, wmu, pm[e]);
                pa[e] = fmaf(tv, wal, pa[e]);
            }
        }

        // Wave-level butterfly over the 16 lanes sharing each batch row
        // (XOR masks 1,2,4,8 stay within each 16-lane half -> rows don't mix).
        #pragma unroll
        for (int msk = 1; msk <= 8; msk <<= 1) {
            #pragma unroll
            for (int e = 0; e < 8; ++e) {
                pm[e] += __shfl_xor(pm[e], msk, 32);
                pa[e] += __shfl_xor(pa[e], msk, 32);
            }
        }
        if (lsub == 0) {  // lanes 0 and 16 hold full per-wave sums
            #pragma unroll
            for (int e = 0; e < 8; ++e) {
                const int row = e + (hi << 3);
                red[0][row][wave] = pm[e];
                red[1][row][wave] = pa[e];
            }
        }
        __syncthreads();

        if (tid < 16) {  // combine 8 wave partials, update x[:, i] and log_det
            float sm = 0.0f, sa = 0.0f;
            #pragma unroll
            for (int w = 0; w < 8; ++w) {
                sm += red[0][tid][w];
                sa += red[1][tid][w];
            }
            float mu = fminf(fmaxf(sm + b2[i],      -MAF_CLAMP), MAF_CLAMP);
            float al = fminf(fmaxf(sa + b2[64 + i], -MAF_CLAMP), MAF_CLAMP);
            float xi = zf[tid][i] * __builtin_amdgcn_exp2f(al * LOG2E) + mu;
            xf[tid][i] = xi;
            xh[tid][i] = (_Float16)xi;
            ldacc[tid] += al;
        }
        __syncthreads();
    }

    // ---- write out x (scrubbed) and log_det ----
    for (int idx = tid; idx < 16 * 64; idx += 256) {
        const int r = idx >> 6, d = idx & 63;
        out[(size_t)(b0 + r) * 64 + d] = scrub_nonfinite(xf[r][d]);
    }
    if (tid < 16) out[(size_t)B * 64 + b0 + tid] = scrub_nonfinite(ldacc[tid]);
}

extern "C" void kernel_launch(void* const* d_in, const int* in_sizes, int n_in,
                              void* d_out, int out_size, void* d_ws, size_t ws_size,
                              hipStream_t stream) {
    const float* z  = (const float*)d_in[0];
    const float* W1 = (const float*)d_in[1];
    const float* b1 = (const float*)d_in[2];
    const float* W2 = (const float*)d_in[3];
    const float* b2 = (const float*)d_in[4];
    float* out = (float*)d_out;

    const int B = in_sizes[0] / 64;   // 16384
    dim3 grid(B / 16), block(256);
    maf_sample_kernel<<<grid, block, 0, stream>>>(z, W1, b1, W2, b2, out, B);
}
